// InvariantPointAttention_33225867002414
// MI455X (gfx1250) — matrix operand order repaired and checked by reading, not compiled, not run on hardware
//
#include <hip/hip_runtime.h>
#include <hip/hip_bf16.h>
#include <math.h>

typedef __attribute__((ext_vector_type(16))) __bf16 v16bf;
typedef __attribute__((ext_vector_type(8)))  float  v8f;

#define NRES 768
#define NH   12
#define CC   16
#define CZ_  128

// ---------------------------------------------------------------------------
// Generic GEMM: C[M x N] = A[M x K] @ B[K x N] + bias[N], bf16 WMMA, fp32 acc.
// 8 waves per block, one 16x16 output tile per wave. M mult of 16, K mult of
// 64. Out-of-range N columns: load clamped garbage (column n of B only feeds
// output column n, which is never stored when OOB) -> zero branches.
// ---------------------------------------------------------------------------
__global__ __launch_bounds__(256) void gemm_wmma_bf16(
    const float* __restrict__ A, const float* __restrict__ B,
    const float* __restrict__ bias, float* __restrict__ C,
    int M, int N, int K, int ldc)
{
  const int lane = threadIdx.x & 31;
  const int wave = threadIdx.x >> 5;
  const int half = lane >> 4;
  const int lcol = lane & 15;
  const int MT = M >> 4;
  const int NT = (N + 15) >> 4;
  const int tile = blockIdx.x * 8 + wave;
  if (tile >= MT * NT) return;           // wave-uniform exit
  const int mt = tile / NT, nt = tile % NT;
  const int m0 = mt * 16, n0 = nt * 16;

  const int arow = m0 + lcol;
  const int bcol = n0 + lcol;
  const bool colok = (bcol < N);
  const float* Abase = A + (size_t)arow * K + half * 16;
  // clamped column: OOB lanes produce garbage in a column we never store
  const float* Bbase = B + (size_t)(half * 16) * N + (colok ? bcol : (N - 1));

  v8f acc = {};
  for (int k0 = 0; k0 < K; k0 += 64) {
    v16bf af0, af1, bf0, bf1;
    const float* Ar0 = Abase + k0;
    const float* Ar1 = Ar0 + 32;
#pragma unroll
    for (int e = 0; e < 16; ++e) {
      const int kk = e;  // {0..15} (+16 for half 1)
      af0[e] = (__bf16)Ar0[kk];
      af1[e] = (__bf16)Ar1[kk];
    }
    const float* Br0 = Bbase + (size_t)k0 * N;
    const float* Br1 = Bbase + (size_t)(k0 + 32) * N;
#pragma unroll
    for (int e = 0; e < 16; ++e) {
      bf0[e] = (__bf16)(*Br0);
      bf1[e] = (__bf16)(*Br1);
      Br0 += N;
      Br1 += N;
    }
    acc = __builtin_amdgcn_wmma_f32_16x16x32_bf16(false, af0, false, bf0,
                                                  (short)0, acc, false, false);
    acc = __builtin_amdgcn_wmma_f32_16x16x32_bf16(false, af1, false, bf1,
                                                  (short)0, acc, false, false);
  }

  if (colok) {
    const float bv = bias ? bias[bcol] : 0.0f;
#pragma unroll
    for (int r = 0; r < 8; ++r) {
      const int crow = m0 + r + half * 8;
      C[(size_t)crow * ldc + bcol] = acc[r] + bv;
    }
  }
}

// ---------------------------------------------------------------------------
// Rearrange projections into per-head-contiguous layouts: q,k,v : [H][N][C]
// ---------------------------------------------------------------------------
__global__ void rearrange_qkv(const float* __restrict__ tq,
                              const float* __restrict__ tkv,
                              float* __restrict__ q, float* __restrict__ k,
                              float* __restrict__ v)
{
  int idx = blockIdx.x * blockDim.x + threadIdx.x;
  if (idx >= NRES * NH * CC) return;
  int c = idx & 15;
  int h = (idx >> 4) % NH;
  int n = idx / (NH * CC);
  size_t dst = ((size_t)h * NRES + n) * CC + c;
  q[dst] = tq[(size_t)n * (NH * CC) + h * CC + c];
  k[dst] = tkv[(size_t)n * (NH * 2 * CC) + h * 2 * CC + c];
  v[dst] = tkv[(size_t)n * (NH * 2 * CC) + h * 2 * CC + CC + c];
}

// ---------------------------------------------------------------------------
// Apply rigid frames to projected points.
//   tqp  : [N][3*48]  -> qp [N][H][4][3] (world frame)
//   tkvp : [N][3*144] -> kp [N][H][4][3], vp [N][H][8][3]
// ---------------------------------------------------------------------------
__global__ __launch_bounds__(192) void ipa_points(
    const float* __restrict__ tqp, const float* __restrict__ tkvp,
    const float* __restrict__ rot, const float* __restrict__ trans,
    float* __restrict__ qp, float* __restrict__ kp, float* __restrict__ vp)
{
  int n = blockIdx.x;
  int t = threadIdx.x;
  __shared__ float R[9], T[3];
  if (t < 9) R[t] = rot[(size_t)n * 9 + t];
  if (t < 3) T[t] = trans[(size_t)n * 3 + t];
  __syncthreads();

  if (t < 48) {  // query points: t = h*PQ + p
    float l0 = tqp[(size_t)n * 144 + 0 * 48 + t];
    float l1 = tqp[(size_t)n * 144 + 1 * 48 + t];
    float l2 = tqp[(size_t)n * 144 + 2 * 48 + t];
    float wx = R[0] * l0 + R[1] * l1 + R[2] * l2 + T[0];
    float wy = R[3] * l0 + R[4] * l1 + R[5] * l2 + T[1];
    float wz = R[6] * l0 + R[7] * l1 + R[8] * l2 + T[2];
    float* dst = qp + ((size_t)n * 48 + t) * 3;
    dst[0] = wx; dst[1] = wy; dst[2] = wz;
  }
  if (t < 144) {  // kv points: t = h*12 + pp
    float l0 = tkvp[(size_t)n * 432 + 0 * 144 + t];
    float l1 = tkvp[(size_t)n * 432 + 1 * 144 + t];
    float l2 = tkvp[(size_t)n * 432 + 2 * 144 + t];
    float wx = R[0] * l0 + R[1] * l1 + R[2] * l2 + T[0];
    float wy = R[3] * l0 + R[4] * l1 + R[5] * l2 + T[1];
    float wz = R[6] * l0 + R[7] * l1 + R[8] * l2 + T[2];
    int h = t / 12, pp = t % 12;
    float* dst = (pp < 4)
        ? kp + (((size_t)n * NH + h) * 4 + pp) * 3
        : vp + (((size_t)n * NH + h) * 8 + (pp - 4)) * 3;
    dst[0] = wx; dst[1] = wy; dst[2] = wz;
  }
}

// ---------------------------------------------------------------------------
// Fused attention: one block (8 wave32) per residue i.
// ---------------------------------------------------------------------------
__global__ __launch_bounds__(256) void ipa_attention(
    const float* __restrict__ q,   // [H][N][C]
    const float* __restrict__ k,   // [H][N][C]
    const float* __restrict__ v,   // [H][N][C]
    const float* __restrict__ qp,  // [N][H][4][3]
    const float* __restrict__ kp,  // [N][H][4][3]
    const float* __restrict__ vp,  // [N][H][8][3]
    const float* __restrict__ bias,// [N][N][12]
    const float* __restrict__ z,   // [N][N][128]
    const float* __restrict__ rot, const float* __restrict__ trans,
    const float* __restrict__ mask, const float* __restrict__ head_w,
    float* __restrict__ cat)       // [N][2112]
{
  __shared__ float  l_logits[NH][NRES];   // probs (f32) after softmax
  __shared__ __bf16 l_probs[16][NRES];    // padded WMMA A-matrix
  __shared__ float  l_q[NH][CC];
  __shared__ float  l_qp[NH][4][3];
  __shared__ float  l_hw[NH];
  __shared__ float  l_acc[NH][40];        // [h][0..15]=o, [h][16+p*3+x]=o_pt

  const int i    = blockIdx.x;
  const int tid  = threadIdx.x;
  const int wave = tid >> 5;
  const int lane = tid & 31;

  for (int t = tid; t < NH * CC; t += 256) {
    int h = t >> 4, c = t & 15;
    l_q[h][c] = q[((size_t)h * NRES + i) * CC + c];
  }
  for (int t = tid; t < NH * 12; t += 256) {
    int h = t / 12, r = t % 12;
    l_qp[h][r / 3][r % 3] = qp[(((size_t)i * NH + h) * 4 + r / 3) * 3 + (r % 3)];
  }
  if (tid < NH) {
    float w = head_w[tid];
    float sp = (w > 20.f) ? w : log1pf(__expf(w));
    l_hw[tid] = sp * 0.13608276348795434f;  // sqrt(1/54)
  }
  for (int t = tid; t < 4 * NRES; t += 256)
    l_probs[12 + t / NRES][t % NRES] = (__bf16)0.0f;
  __syncthreads();

  const float mi = mask[i];

  // ---- pass1: logits ----
  for (int j = tid; j < NRES; j += 256) {
    float msk = 100000.0f * (mi * mask[j] - 1.0f);
    float brow[12];
    {
      const float4* bz = (const float4*)(bias + ((size_t)i * NRES + j) * NH);
      float4 b0 = bz[0], b1 = bz[1], b2 = bz[2];
      brow[0] = b0.x; brow[1] = b0.y; brow[2]  = b0.z; brow[3]  = b0.w;
      brow[4] = b1.x; brow[5] = b1.y; brow[6]  = b1.z; brow[7]  = b1.w;
      brow[8] = b2.x; brow[9] = b2.y; brow[10] = b2.z; brow[11] = b2.w;
    }
#pragma unroll 1
    for (int h = 0; h < NH; ++h) {
      const float* kr = k + ((size_t)h * NRES + j) * CC;
      float acc = 0.f;
#pragma unroll
      for (int c = 0; c < CC; ++c) acc += l_q[h][c] * kr[c];
      acc *= 0.14433756729740643f;  // sqrt(1/(3*16))
      acc += 0.5773502691896258f * brow[h];
      const float* kpr = kp + ((size_t)j * NH + h) * 12;
      float pt = 0.f;
#pragma unroll
      for (int p = 0; p < 4; ++p) {
        float dx = l_qp[h][p][0] - kpr[p * 3 + 0];
        float dy = l_qp[h][p][1] - kpr[p * 3 + 1];
        float dz = l_qp[h][p][2] - kpr[p * 3 + 2];
        pt += dx * dx + dy * dy + dz * dz;
      }
      l_logits[h][j] = acc - 0.5f * l_hw[h] * pt + msk;
    }
  }
  __syncthreads();

  // ---- pass2: softmax per head ----
  for (int h = wave; h < NH; h += 8) {
    float mx = -3.0e38f;
    for (int j = lane; j < NRES; j += 32) mx = fmaxf(mx, l_logits[h][j]);
#pragma unroll
    for (int off = 16; off; off >>= 1) mx = fmaxf(mx, __shfl_xor(mx, off, 32));
    float sum = 0.f;
    for (int j = lane; j < NRES; j += 32) {
      float e = __expf(l_logits[h][j] - mx);
      l_logits[h][j] = e;
      sum += e;
    }
#pragma unroll
    for (int off = 16; off; off >>= 1) sum += __shfl_xor(sum, off, 32);
    float inv = 1.0f / sum;
    for (int j = lane; j < NRES; j += 32) {
      float p = l_logits[h][j] * inv;
      l_logits[h][j] = p;
      l_probs[h][j]  = (__bf16)p;
    }
  }
  __syncthreads();

  // ---- pass3a: o and o_pt (480 outputs, <=2 per thread) ----
  {
    int idx0 = tid, idx1 = tid + 256;
    int h0 = idx0 / 40, r0 = idx0 % 40;
    const float* p0 = (r0 < CC)
        ? v  + (size_t)h0 * NRES * CC + r0
        : vp + (size_t)h0 * 24 + (r0 - CC);
    size_t s0 = (r0 < CC) ? (size_t)CC : (size_t)(NH * 24);
    bool has1 = idx1 < 480;
    int h1 = has1 ? idx1 / 40 : 0, r1 = has1 ? idx1 % 40 : 0;
    const float* p1 = (r1 < CC)
        ? v  + (size_t)h1 * NRES * CC + r1
        : vp + (size_t)h1 * 24 + (r1 - CC);
    size_t s1 = (r1 < CC) ? (size_t)CC : (size_t)(NH * 24);
    float a0 = 0.f, a1 = 0.f;
    for (int j = 0; j < NRES; ++j) {
      a0 += l_logits[h0][j] * p0[(size_t)j * s0];
      if (has1) a1 += l_logits[h1][j] * p1[(size_t)j * s1];
    }
    l_acc[h0][r0] = a0;
    if (has1) l_acc[h1][r1] = a1;
  }

  // ---- pass3b: o_pair via WMMA; wave w owns columns [w*16, w*16+16) ----
  {
    const int n0   = wave * 16;
    const int lcol = lane & 15;
    const int half = lane >> 4;
    v8f acc8 = {};
    const size_t zbase = (size_t)i * NRES * CZ_;
    const __bf16* Abase = &l_probs[lcol][half * 16];
    const float*  Bbase = z + zbase + (size_t)(half * 16) * CZ_ + (n0 + lcol);
    for (int k0 = 0; k0 < NRES; k0 += 64) {
      v16bf af0, af1, bf0, bf1;
      const __bf16* ar0 = Abase + k0;
      const __bf16* ar1 = ar0 + 32;
#pragma unroll
      for (int e = 0; e < 16; ++e) {
        const int kk = e;
        af0[e] = ar0[kk];
        af1[e] = ar1[kk];
      }
      const float* br0 = Bbase + (size_t)k0 * CZ_;
      const float* br1 = Bbase + (size_t)(k0 + 32) * CZ_;
#pragma unroll
      for (int e = 0; e < 16; ++e) {
        bf0[e] = (__bf16)__builtin_nontemporal_load(br0);  // z streamed once
        bf1[e] = (__bf16)__builtin_nontemporal_load(br1);
        br0 += CZ_;
        br1 += CZ_;
      }
      acc8 = __builtin_amdgcn_wmma_f32_16x16x32_bf16(false, af0, false, bf0,
                                                     (short)0, acc8, false, false);
      acc8 = __builtin_amdgcn_wmma_f32_16x16x32_bf16(false, af1, false, bf1,
                                                     (short)0, acc8, false, false);
    }
    float* crow = cat + (size_t)i * 2112 + 576;
#pragma unroll
    for (int r = 0; r < 8; ++r) {
      int h = r + half * 8;
      if (h < NH) crow[h * CZ_ + n0 + lcol] = acc8[r];
    }
  }
  __syncthreads();

  // ---- pass4: finalize ----
  float* base = cat + (size_t)i * 2112;
  if (tid < NH * CC) {
    int h = tid >> 4, c = tid & 15;
    base[h * CC + c] = l_acc[h][c];  // o region
  }
  if (tid >= 192 && tid < 192 + NH * 8) {
    int u = tid - 192;
    int h = u / 8, p2 = u % 8;
    float gx = l_acc[h][16 + p2 * 3 + 0] - trans[(size_t)i * 3 + 0];
    float gy = l_acc[h][16 + p2 * 3 + 1] - trans[(size_t)i * 3 + 1];
    float gz = l_acc[h][16 + p2 * 3 + 2] - trans[(size_t)i * 3 + 2];
    const float* R = rot + (size_t)i * 9;
    float lx = R[0] * gx + R[3] * gy + R[6] * gz;  // R^T * g
    float ly = R[1] * gx + R[4] * gy + R[7] * gz;
    float lz = R[2] * gx + R[5] * gy + R[8] * gz;
    float nrm = sqrtf(lx * lx + ly * ly + lz * lz + 1e-8f);
    int o = h * 8 + p2;
    base[192 + o] = lx;
    base[288 + o] = ly;
    base[384 + o] = lz;
    base[480 + o] = nrm;
  }
}

// ---------------------------------------------------------------------------
extern "C" void kernel_launch(void* const* d_in, const int* in_sizes, int n_in,
                              void* d_out, int out_size, void* d_ws, size_t ws_size,
                              hipStream_t stream)
{
  (void)in_sizes; (void)n_in; (void)out_size; (void)ws_size;
  const float* s      = (const float*)d_in[0];
  const float* z      = (const float*)d_in[1];
  const float* rot    = (const float*)d_in[2];
  const float* trans  = (const float*)d_in[3];
  const float* maskp  = (const float*)d_in[4];
  const float* Wq     = (const float*)d_in[5];
  const float* bq     = (const float*)d_in[6];
  const float* Wkv    = (const float*)d_in[7];
  const float* bkv    = (const float*)d_in[8];
  const float* Wqp    = (const float*)d_in[9];
  const float* bqp    = (const float*)d_in[10];
  const float* Wkvp   = (const float*)d_in[11];
  const float* bkvp   = (const float*)d_in[12];
  const float* Wb     = (const float*)d_in[13];
  const float* bb     = (const float*)d_in[14];
  const float* head_w = (const float*)d_in[15];
  const float* Wout   = (const float*)d_in[16];
  const float* bout   = (const float*)d_in[17];
  float* out = (float*)d_out;

  // workspace partition (floats)
  float* ws   = (float*)d_ws;
  float* tq   = ws;                    // 768*192
  float* tkv  = tq   + 768 * 192;      // 768*384
  float* tqp  = tkv  + 768 * 384;      // 768*144
  float* tkvp = tqp  + 768 * 144;      // 768*432
  float* qb   = tkvp + 768 * 432;      // 12*768*16
  float* kb   = qb   + NH * NRES * CC;
  float* vb   = kb   + NH * NRES * CC;
  float* qpb  = vb   + NH * NRES * CC; // 768*12*4*3
  float* kpb  = qpb  + NRES * NH * 12;
  float* vpb  = kpb  + NRES * NH * 12; // 768*12*8*3
  float* bbf  = vpb  + NRES * NH * 24; // 768*768*12
  float* catb = bbf  + (size_t)NRES * NRES * NH; // 768*2112

  dim3 blk(256);
  auto tiles = [](int M, int N) { return ((M >> 4) * ((N + 15) >> 4) + 7) / 8; };
  // projections from s
  gemm_wmma_bf16<<<dim3(tiles(768, 192)), blk, 0, stream>>>(s, Wq,   bq,   tq,   768, 192, 384, 192);
  gemm_wmma_bf16<<<dim3(tiles(768, 384)), blk, 0, stream>>>(s, Wkv,  bkv,  tkv,  768, 384, 384, 384);
  gemm_wmma_bf16<<<dim3(tiles(768, 144)), blk, 0, stream>>>(s, Wqp,  bqp,  tqp,  768, 144, 384, 144);
  gemm_wmma_bf16<<<dim3(tiles(768, 432)), blk, 0, stream>>>(s, Wkvp, bkvp, tkvp, 768, 432, 384, 432);
  // pair bias: (N*N) x 128 @ 128 x 12
  gemm_wmma_bf16<<<dim3(tiles(NRES * NRES, NH)), blk, 0, stream>>>(
      z, Wb, bb, bbf, NRES * NRES, NH, CZ_, NH);
  // layout transforms
  rearrange_qkv<<<dim3((NRES * NH * CC + 255) / 256), blk, 0, stream>>>(
      tq, tkv, qb, kb, vb);
  ipa_points<<<dim3(NRES), dim3(192), 0, stream>>>(tqp, tkvp, rot, trans,
                                                   qpb, kpb, vpb);
  // fused attention
  ipa_attention<<<dim3(NRES), blk, 0, stream>>>(
      qb, kb, vb, qpb, kpb, vpb, bbf, z, rot, trans, maskp, head_w, catb);
  // output projection
  gemm_wmma_bf16<<<dim3(tiles(768, 384)), blk, 0, stream>>>(
      catb, Wout, bout, out, 768, 384, 2112, 384);
}
